// GNN4CD_Model_64321430224925
// MI455X (gfx1250) — matrix operand order repaired and checked by reading, not compiled, not run on hardware
//
#include <hip/hip_runtime.h>

#define N_LOW      20000
#define N_LOW_PAD  20032
#define N_HIGH     100000
#define N_HIGH_PAD 100032
#define E_L2H_N    400000
#define E_HH_N     600000
#define F_LOW_N    28
#define ENC_D      128
#define ROWT       4

typedef __attribute__((ext_vector_type(16))) __bf16 v16bf;
typedef __attribute__((ext_vector_type(8)))  float  v8f;

union BFVec { v16bf v; unsigned short u[16]; };

__device__ __forceinline__ unsigned short f2bf(float x) {
  unsigned u = __float_as_uint(x);
  u += 0x7FFFu + ((u >> 16) & 1u);   // round-to-nearest-even to bf16
  return (unsigned short)(u >> 16);
}

// ordered-uint mapping for float atomicMax
__device__ __forceinline__ unsigned f2ord(float f) {
  unsigned u = __float_as_uint(f);
  return (u & 0x80000000u) ? ~u : (u | 0x80000000u);
}
__device__ __forceinline__ float ord2f(unsigned u) {
  unsigned v = (u & 0x80000000u) ? (u & 0x7FFFFFFFu) : ~u;
  return __uint_as_float(v);
}

// ---------------------------------------------------------------------------
// Weight pre-pack into per-lane WMMA B-fragment order, converted to bf16.
// B as packed here (this code's k order, same as the A reads): lane n = ct*16 + (lane&15); lanes 0-15 hold
// K = kb..kb+15, lanes 16-31 hold K = kb+16..kb+31, contiguous over 8 VGPRs.
// ---------------------------------------------------------------------------
__global__ void prepack_w(const float* __restrict__ W, unsigned short* __restrict__ Wpk,
                          int Kreal, int Nout, int NT, int total)
{
  int t = blockIdx.x * blockDim.x + threadIdx.x;
  if (t >= total) return;
  int lane = t & 31;
  int ct   = (t >> 5) % NT;
  int kb5  = (t >> 5) / NT;
  int half = lane >> 4;
  int n    = ct * 16 + (lane & 15);
  unsigned short* dst = Wpk + ((size_t)t << 4);
#pragma unroll
  for (int i = 0; i < 16; ++i) {
    int k = (kb5 << 5) + (half << 4) + i;
    float v = (n < Nout && k < Kreal) ? W[(size_t)k * Nout + n] : 0.f;
    dst[i] = f2bf(v);
  }
}

// zero-pad rows: dst[n,0:Kout] = src[n,0:Kin] || 0
__global__ void pad_rows(const float* __restrict__ src, float* __restrict__ dst,
                         int N, int Kin, int Kout)
{
  long long t = (long long)blockIdx.x * blockDim.x + threadIdx.x;
  if (t >= (long long)N * Kout) return;
  int n = (int)(t / Kout), k = (int)(t % Kout);
  dst[t] = (k < Kin) ? src[(long long)n * Kin + k] : 0.f;
}

// ---------------------------------------------------------------------------
// WMMA GEMM: Out[.,Nout] = A[.,Kpad] @ W + bias (optional relu)
// Requires: Kpad%32==0, Mpad%64==0, A readable for Mpad rows (padded bufs).
// Block: 32*NT threads; wave = one 16-col tile; 4 row tiles (64 rows)/block.
// A tile staged f32->bf16 in LDS cooperatively; fragments via ds_load_b128.
// ---------------------------------------------------------------------------
__global__ void __launch_bounds__(256) wmma_gemm(
    const float* __restrict__ A, const unsigned short* __restrict__ Wpk,
    const float* __restrict__ bias, float* __restrict__ Out,
    int Mstore, int Kpad, int Nout, int doRelu)
{
  __shared__ unsigned short lds[64 * 32];     // 64 rows x 32 K, bf16
  const int NT   = blockDim.x >> 5;
  const int wave = threadIdx.x >> 5;
  const int lane = threadIdx.x & 31;
  const int half = lane >> 4;
  const int mr   = lane & 15;
  const int row0 = blockIdx.x * 64;
  const int KB   = Kpad >> 5;

  v8f acc[ROWT];
#pragma unroll
  for (int r = 0; r < ROWT; ++r) acc[r] = (v8f){0.f,0.f,0.f,0.f,0.f,0.f,0.f,0.f};

  for (int kb5 = 0; kb5 < KB; ++kb5) {
    // ---- cooperative stage: A[row0..row0+63, kb..kb+31] -> LDS bf16 ----
    for (int c = threadIdx.x; c < 512; c += blockDim.x) {
      int r = c >> 3, seg = c & 7;           // row, 4-float segment
      const float4 f = *(const float4*)(A + (size_t)(row0 + r) * Kpad + (kb5 << 5) + (seg << 2));
      unsigned long long pkv =  (unsigned long long)f2bf(f.x)
                             | ((unsigned long long)f2bf(f.y) << 16)
                             | ((unsigned long long)f2bf(f.z) << 32)
                             | ((unsigned long long)f2bf(f.w) << 48);
      *(unsigned long long*)(&lds[r * 32 + (seg << 2)]) = pkv;
    }
    __syncthreads();

    BFVec b;
    b.v = *(const v16bf*)(Wpk + ((((size_t)kb5 * NT + wave) * 32 + lane) << 4));
#pragma unroll
    for (int rt = 0; rt < ROWT; ++rt) {
      const int r = rt * 16 + mr;
      union { uint4 q[2]; BFVec a; } af;
      af.q[0] = *(const uint4*)(&lds[r * 32 + (half << 4)]);
      af.q[1] = *(const uint4*)(&lds[r * 32 + (half << 4) + 8]);
      acc[rt] = __builtin_amdgcn_wmma_f32_16x16x32_bf16(false, af.a.v, false, b.v,
                                                        (short)0, acc[rt], false, false);
    }
    __syncthreads();
  }

  const int ocol = (wave << 4) + mr;
  if (ocol < Nout) {
    const float bb = bias ? bias[ocol] : 0.f;
#pragma unroll
    for (int rt = 0; rt < ROWT; ++rt) {
      const int rb = row0 + rt * 16 + (half << 3);
#pragma unroll
      for (int r = 0; r < 8; ++r) {
        int orow = rb + r;
        if (orow < Mstore) {
          float v = acc[rt][r] + bb;
          if (doRelu) v = fmaxf(v, 0.f);
          Out[(size_t)orow * Nout + ocol] = v;
        }
      }
    }
  }
}

// ---------------------------------------------------------------------------
// Utility fills
// ---------------------------------------------------------------------------
__global__ void fill_f32(float* p, float v, long long n) {
  long long i = (long long)blockIdx.x * blockDim.x + threadIdx.x;
  if (i < n) p[i] = v;
}
__global__ void fill_u32(unsigned* p, unsigned v, long long n) {
  long long i = (long long)blockIdx.x * blockDim.x + threadIdx.x;
  if (i < n) p[i] = v;
}

// ---------------------------------------------------------------------------
// Bipartite GraphConv (low->high, mean): wave per edge, 128 channels
// ---------------------------------------------------------------------------
__global__ void l2h_edge(const float* __restrict__ enc, const int* __restrict__ src,
                         const int* __restrict__ dst, float* __restrict__ agg,
                         float* __restrict__ cnt, int E)
{
  long long t = (long long)blockIdx.x * blockDim.x + threadIdx.x;
  int e = (int)(t >> 5);
  int lane = threadIdx.x & 31;
  if (e >= E) return;
  int s = src[e], d = dst[e];
  const float* row = enc + (long long)s * 128;
  float* out = agg + (long long)d * 128;
#pragma unroll
  for (int i = 0; i < 4; ++i) atomicAdd(&out[lane * 4 + i], row[lane * 4 + i]);
  if (lane == 0) atomicAdd(&cnt[d], 1.f);
}

__global__ void l2h_div(float* __restrict__ agg, const float* __restrict__ cnt, int N)
{
  long long t = (long long)blockIdx.x * blockDim.x + threadIdx.x;
  if (t >= (long long)N * 128) return;
  agg[t] /= fmaxf(cnt[t >> 7], 1.f);
}

// x[n,j] += sum_k x_high[n,k] * Wroot[k,j]  (K=4 root term of GraphConv)
__global__ void add_root(float* __restrict__ x, const float* __restrict__ xh,
                         const float* __restrict__ Wroot, int N)
{
  long long t = (long long)blockIdx.x * blockDim.x + threadIdx.x;
  if (t >= (long long)N * 64) return;
  int nn = (int)(t >> 6), j = (int)(t & 63);
  float acc = x[t];
#pragma unroll
  for (int k = 0; k < 4; ++k) acc += xh[nn * 4 + k] * Wroot[k * 64 + j];
  x[t] = acc;
}

// ---------------------------------------------------------------------------
// BatchNorm (biased batch stats), in place, optional relu. One block/channel.
// ---------------------------------------------------------------------------
__global__ void bn_kernel(float* __restrict__ x, const float* __restrict__ g,
                          const float* __restrict__ b, int N, int C, int doRelu)
{
  const int ch = blockIdx.x;
  __shared__ float s1[256], s2[256];
  float sum = 0.f, sq = 0.f;
  for (int i = threadIdx.x; i < N; i += blockDim.x) {
    float v = x[(long long)i * C + ch];
    sum += v; sq += v * v;
  }
  s1[threadIdx.x] = sum; s2[threadIdx.x] = sq;
  __syncthreads();
  for (int s = 128; s > 0; s >>= 1) {
    if ((int)threadIdx.x < s) { s1[threadIdx.x] += s1[threadIdx.x + s]; s2[threadIdx.x] += s2[threadIdx.x + s]; }
    __syncthreads();
  }
  float mu = s1[0] / (float)N;
  float var = s2[0] / (float)N - mu * mu;
  float scale = g[ch] * rsqrtf(var + 1e-5f);
  float off = b[ch] - mu * scale;
  for (int i = threadIdx.x; i < N; i += blockDim.x) {
    float v = x[(long long)i * C + ch] * scale + off;
    if (doRelu) v = fmaxf(v, 0.f);
    x[(long long)i * C + ch] = v;
  }
}

// ---------------------------------------------------------------------------
// GATv2 edge kernels. Virtual edges [0,E) = real, [E,E+N) = self loops.
// ---------------------------------------------------------------------------
__global__ void gat_logits(const float* __restrict__ xl, const float* __restrict__ xr,
                           const int* __restrict__ src, const int* __restrict__ dst,
                           const float* __restrict__ att, float* __restrict__ logit,
                           unsigned* __restrict__ mmax, int E, int N, int HC, int H)
{
  long long t = (long long)blockIdx.x * blockDim.x + threadIdx.x;
  int e = (int)(t >> 5);
  int lane = threadIdx.x & 31;
  if (e >= E + N) return;
  int s = (e < E) ? src[e] : (e - E);
  int d = (e < E) ? dst[e] : (e - E);
  const int epl = HC >> 5;                 // channels per lane (4 or 2)
  const float* pl = xl + (long long)s * HC + lane * epl;
  const float* pr = xr + (long long)d * HC + lane * epl;
  float partial = 0.f;
  for (int i = 0; i < epl; ++i) {
    float v = pl[i] + pr[i];
    v = (v > 0.f) ? v : 0.2f * v;          // leaky_relu(0.2)
    partial += att[lane * epl + i] * v;
  }
  for (int off = 1; off < 16; off <<= 1) partial += __shfl_xor(partial, off, 32);
  if (H == 1) {
    partial += __shfl_xor(partial, 16, 32);
    if (lane == 0) {
      logit[e] = partial;
      atomicMax(&mmax[d], f2ord(partial));
    }
  } else {
    if ((lane & 15) == 0) {
      int h = lane >> 4;
      logit[(long long)e * 2 + h] = partial;
      atomicMax(&mmax[(long long)d * 2 + h], f2ord(partial));
    }
  }
}

__global__ void gat_exp(float* __restrict__ w, const unsigned* __restrict__ mmax,
                        const int* __restrict__ dst, float* __restrict__ z,
                        int E, int N, int H)
{
  long long t = (long long)blockIdx.x * blockDim.x + threadIdx.x;
  if (t >= (long long)(E + N) * H) return;
  int e = (int)(t / H), h = (int)(t % H);
  int d = (e < E) ? dst[e] : (e - E);
  float m = ord2f(mmax[(long long)d * H + h]);
  float val = __expf(w[t] - m);
  w[t] = val;
  atomicAdd(&z[(long long)d * H + h], val);
}

__global__ void gat_agg(const float* __restrict__ xl, const float* __restrict__ w,
                        const float* __restrict__ z, const int* __restrict__ src,
                        const int* __restrict__ dst, float* __restrict__ agg,
                        int E, int N, int HC, int H)
{
  long long t = (long long)blockIdx.x * blockDim.x + threadIdx.x;
  int e = (int)(t >> 5);
  int lane = threadIdx.x & 31;
  if (e >= E + N) return;
  int s = (e < E) ? src[e] : (e - E);
  int d = (e < E) ? dst[e] : (e - E);
  const int epl = HC >> 5;
  const int C = HC / H;
  const float* pl = xl + (long long)s * HC;
  float* pa = agg + (long long)d * HC;
  for (int i = 0; i < epl; ++i) {
    int ch = lane * epl + i;
    int h = ch / C;
    float coef = w[(long long)e * H + h] / z[(long long)d * H + h];
    atomicAdd(&pa[ch], coef * pl[ch]);
  }
}

__global__ void gat_finalize(float* __restrict__ agg, const float* __restrict__ deg,
                             const float* __restrict__ bias, int N, int HC, int doRelu)
{
  long long t = (long long)blockIdx.x * blockDim.x + threadIdx.x;
  if (t >= (long long)N * HC) return;
  int nn = (int)(t / HC), ch = (int)(t % HC);
  float v = agg[t] / deg[nn] + bias[ch];
  if (doRelu) v = fmaxf(v, 0.f);
  agg[t] = v;
}

__global__ void count_dst(const int* __restrict__ dst, float* __restrict__ deg, int E)
{
  int e = blockIdx.x * blockDim.x + threadIdx.x;
  if (e < E) atomicAdd(&deg[dst[e]], 1.f);
}

// ---------------------------------------------------------------------------
// Host orchestration
// ---------------------------------------------------------------------------
static inline unsigned gdim(long long n, int tpb) { return (unsigned)((n + tpb - 1) / tpb); }

static void launch_gemm(hipStream_t st, const float* A, const unsigned short* Wpk,
                        const float* bias, float* Out, int Mpad, int Mstore,
                        int Kpad, int Nout, int NoutPad, int relu)
{
  wmma_gemm<<<Mpad / 64, 32 * (NoutPad >> 4), 0, st>>>(A, Wpk, bias, Out, Mstore, Kpad, Nout, relu);
}

static void run_gat(hipStream_t st, const float* x_in, int din, int H, int C,
                    const unsigned short* pkWl, const float* bl,
                    const unsigned short* pkWr, const float* br,
                    const float* att, const float* bias,
                    float* xl, float* xr, float* out, float* wbuf, unsigned* mmax,
                    float* zbuf, const float* deg, const int* src, const int* dst,
                    int E, int N, int relu)
{
  const int HC = H * C;
  launch_gemm(st, x_in, pkWl, bl, xl, N_HIGH_PAD, N_HIGH_PAD, din, HC, HC, 0);
  launch_gemm(st, x_in, pkWr, br, xr, N_HIGH_PAD, N_HIGH_PAD, din, HC, HC, 0);
  fill_u32<<<gdim((long long)N * H, 256), 256, 0, st>>>(mmax, 0u, (long long)N * H);
  fill_f32<<<gdim((long long)N * H, 256), 256, 0, st>>>(zbuf, 0.f, (long long)N * H);
  fill_f32<<<gdim((long long)N * HC, 256), 256, 0, st>>>(out, 0.f, (long long)N * HC);
  long long Etot = (long long)E + N;
  gat_logits<<<gdim(Etot * 32, 256), 256, 0, st>>>(xl, xr, src, dst, att, wbuf, mmax, E, N, HC, H);
  gat_exp<<<gdim(Etot * H, 256), 256, 0, st>>>(wbuf, mmax, dst, zbuf, E, N, H);
  gat_agg<<<gdim(Etot * 32, 256), 256, 0, st>>>(xl, wbuf, zbuf, src, dst, out, E, N, HC, H);
  gat_finalize<<<gdim((long long)N * HC, 256), 256, 0, st>>>(out, deg, bias, N, HC, relu);
}

extern "C" void kernel_launch(void* const* d_in, const int* in_sizes, int n_in,
                              void* d_out, int out_size, void* d_ws, size_t ws_size,
                              hipStream_t stream)
{
  // ---- inputs (setup_inputs dict order, params flattened in dict order) ----
  const float* x_low    = (const float*)d_in[0];
  const float* x_high   = (const float*)d_in[1];
  const int*   src_l2h  = (const int*)d_in[2];
  const int*   dst_l2h  = (const int*)d_in[3];
  const int*   src_hh   = (const int*)d_in[4];
  const int*   dst_hh   = (const int*)d_in[5];
  const float* enc_W    = (const float*)d_in[6];
  const float* enc_b    = (const float*)d_in[7];
  const float* dWrel    = (const float*)d_in[8];
  const float* dbrel    = (const float*)d_in[9];
  const float* dWroot   = (const float*)d_in[10];
  const float* bn_g[5], *bn_b[5];
  for (int i = 0; i < 5; ++i) { bn_g[i] = (const float*)d_in[11 + 2 * i]; bn_b[i] = (const float*)d_in[12 + 2 * i]; }
  struct GatP { const float *Wl, *bl, *Wr, *br, *att, *bias; } g[5];
  for (int i = 0; i < 5; ++i) {
    int base = 21 + 6 * i;
    g[i].Wl   = (const float*)d_in[base + 0];
    g[i].bl   = (const float*)d_in[base + 1];
    g[i].Wr   = (const float*)d_in[base + 2];
    g[i].br   = (const float*)d_in[base + 3];
    g[i].att  = (const float*)d_in[base + 4];
    g[i].bias = (const float*)d_in[base + 5];
  }
  const float* p1_W = (const float*)d_in[51];
  const float* p1_b = (const float*)d_in[52];
  const float* p2_W = (const float*)d_in[53];
  const float* p2_b = (const float*)d_in[54];
  const float* p3_W = (const float*)d_in[55];
  const float* p3_b = (const float*)d_in[56];
  float* out = (float*)d_out;

  // ---- workspace carving (all activation buffers padded to M%64==0) ----
  size_t off = 0;
  char* wsbase = (char*)d_ws;
  auto carve = [&](size_t bytes) -> void* {
    void* p = wsbase + off; off += (bytes + 255) & ~(size_t)255; return p;
  };
  float*    xpad = (float*)carve((size_t)N_LOW_PAD * 32 * 4);
  float*    enc  = (float*)carve((size_t)N_LOW_PAD * ENC_D * 4);
  float*    bufA = (float*)carve((size_t)N_HIGH_PAD * 128 * 4);
  float*    bufB = (float*)carve((size_t)N_HIGH_PAD * 128 * 4);
  float*    xl   = (float*)carve((size_t)N_HIGH_PAD * 128 * 4);
  float*    xr   = (float*)carve((size_t)N_HIGH_PAD * 128 * 4);
  float*    wbuf = (float*)carve((size_t)(E_HH_N + N_HIGH) * 2 * 4);
  unsigned* mmax = (unsigned*)carve((size_t)N_HIGH * 2 * 4);
  float*    zbuf = (float*)carve((size_t)N_HIGH * 2 * 4);
  float*    cnt  = (float*)carve((size_t)N_HIGH * 4);
  float*    deg  = (float*)carve((size_t)N_HIGH * 4);
  (void)ws_size; (void)n_in; (void)in_sizes; (void)out_size;

  // ---- weight pre-pack (bf16 fragment layout) ----
  auto pack = [&](const float* W, int Kreal, int Kpad, int Nout, int NoutPad) -> unsigned short* {
    int KB = Kpad >> 5, NT = NoutPad >> 4;
    int total = KB * NT * 32;
    unsigned short* p = (unsigned short*)carve((size_t)total * 32);
    prepack_w<<<gdim(total, 256), 256, 0, stream>>>(W, p, Kreal, Nout, NT, total);
    return p;
  };
  const unsigned short* pk_enc = pack(enc_W, F_LOW_N, 32, 128, 128);
  const unsigned short* pk_rel = pack(dWrel, 128, 128, 64, 64);
  const int din[5] = {64, 128, 128, 128, 128};
  const int HC[5]  = {128, 128, 128, 128, 64};
  const unsigned short *pk_Wl[5], *pk_Wr[5];
  for (int i = 0; i < 5; ++i) {
    pk_Wl[i] = pack(g[i].Wl, din[i], din[i], HC[i], HC[i]);
    pk_Wr[i] = pack(g[i].Wr, din[i], din[i], HC[i], HC[i]);
  }
  const unsigned short* pk_p1 = pack(p1_W, 64, 64, 64, 64);
  const unsigned short* pk_p2 = pack(p2_W, 64, 64, 32, 32);
  const unsigned short* pk_p3 = pack(p3_W, 32, 32, 4, 16);

  // ---- encoder: enc = relu(x_low @ enc_W + enc_b), K padded 28->32 ----
  pad_rows<<<gdim((long long)N_LOW * 32, 256), 256, 0, stream>>>(x_low, xpad, N_LOW, F_LOW_N, 32);
  launch_gemm(stream, xpad, pk_enc, enc_b, enc, N_LOW_PAD, N_LOW_PAD, 32, 128, 128, 1);

  // ---- bipartite GraphConv (mean) low -> high ----
  fill_f32<<<gdim((long long)N_HIGH * 128, 256), 256, 0, stream>>>(bufB, 0.f, (long long)N_HIGH * 128);
  fill_f32<<<gdim(N_HIGH, 256), 256, 0, stream>>>(cnt, 0.f, N_HIGH);
  l2h_edge<<<gdim((long long)E_L2H_N * 32, 256), 256, 0, stream>>>(enc, src_l2h, dst_l2h, bufB, cnt, E_L2H_N);
  l2h_div<<<gdim((long long)N_HIGH * 128, 256), 256, 0, stream>>>(bufB, cnt, N_HIGH);
  launch_gemm(stream, bufB, pk_rel, dbrel, bufA, N_HIGH_PAD, N_HIGH_PAD, 128, 64, 64, 0);
  add_root<<<gdim((long long)N_HIGH * 64, 256), 256, 0, stream>>>(bufA, x_high, dWroot, N_HIGH);

  // ---- bn0 (no relu) ----
  bn_kernel<<<64, 256, 0, stream>>>(bufA, bn_g[0], bn_b[0], N_HIGH, 64, 0);

  // ---- node degrees (self-loop included) ----
  fill_f32<<<gdim(N_HIGH, 256), 256, 0, stream>>>(deg, 1.f, N_HIGH);
  count_dst<<<gdim(E_HH_N, 256), 256, 0, stream>>>(dst_hh, deg, E_HH_N);

  // ---- 5 GATv2 layers ----
  run_gat(stream, bufA, 64, 2, 64, pk_Wl[0], g[0].bl, pk_Wr[0], g[0].br, g[0].att, g[0].bias,
          xl, xr, bufB, wbuf, mmax, zbuf, deg, src_hh, dst_hh, E_HH_N, N_HIGH, 0);
  bn_kernel<<<128, 256, 0, stream>>>(bufB, bn_g[1], bn_b[1], N_HIGH, 128, 1);

  run_gat(stream, bufB, 128, 2, 64, pk_Wl[1], g[1].bl, pk_Wr[1], g[1].br, g[1].att, g[1].bias,
          xl, xr, bufA, wbuf, mmax, zbuf, deg, src_hh, dst_hh, E_HH_N, N_HIGH, 0);
  bn_kernel<<<128, 256, 0, stream>>>(bufA, bn_g[2], bn_b[2], N_HIGH, 128, 1);

  run_gat(stream, bufA, 128, 2, 64, pk_Wl[2], g[2].bl, pk_Wr[2], g[2].br, g[2].att, g[2].bias,
          xl, xr, bufB, wbuf, mmax, zbuf, deg, src_hh, dst_hh, E_HH_N, N_HIGH, 0);
  bn_kernel<<<128, 256, 0, stream>>>(bufB, bn_g[3], bn_b[3], N_HIGH, 128, 1);

  run_gat(stream, bufB, 128, 2, 64, pk_Wl[3], g[3].bl, pk_Wr[3], g[3].br, g[3].att, g[3].bias,
          xl, xr, bufA, wbuf, mmax, zbuf, deg, src_hh, dst_hh, E_HH_N, N_HIGH, 0);
  bn_kernel<<<128, 256, 0, stream>>>(bufA, bn_g[4], bn_b[4], N_HIGH, 128, 1);

  // gat5: heads=1, out=64, relu fused into finalize (no BN after)
  run_gat(stream, bufA, 128, 1, 64, pk_Wl[4], g[4].bl, pk_Wr[4], g[4].br, g[4].att, g[4].bias,
          xl, xr, bufB, wbuf, mmax, zbuf, deg, src_hh, dst_hh, E_HH_N, N_HIGH, 1);

  // ---- MLP head ----
  launch_gemm(stream, bufB, pk_p1, p1_b, bufA, N_HIGH_PAD, N_HIGH_PAD, 64, 64, 64, 1);
  launch_gemm(stream, bufA, pk_p2, p2_b, xl,   N_HIGH_PAD, N_HIGH_PAD, 64, 32, 32, 1);
  launch_gemm(stream, xl,   pk_p3, p3_b, out,  N_HIGH_PAD, N_HIGH,     32, 4, 16, 0);
}
